// HeteroGNN_88940182765819
// MI455X (gfx1250) — matrix operand order repaired and checked by reading, not compiled, not run on hardware
//
#include <hip/hip_runtime.h>
#include <hip/hip_bf16.h>

// ---------------------------------------------------------------------------
// Types for CDNA5 WMMA / TDM
// ---------------------------------------------------------------------------
typedef __bf16 v16bf __attribute__((ext_vector_type(16)));
typedef __bf16 v8bf  __attribute__((ext_vector_type(8)));
typedef float  v8f   __attribute__((ext_vector_type(8)));
typedef unsigned int u32x4 __attribute__((ext_vector_type(4)));
typedef int          i32x4 __attribute__((ext_vector_type(4)));
typedef int          i32x8 __attribute__((ext_vector_type(8)));

union V16U { v16bf v; v8bf h[2]; };

#define HN   128        // hidden size (also N of every GEMM)
#define KC   64         // K chunk staged in LDS

#if __has_builtin(__builtin_amdgcn_tensor_load_to_lds)
#define HAVE_TDM 1
#else
#define HAVE_TDM 0
#endif

// ---------------------------------------------------------------------------
// Generic fill
// ---------------------------------------------------------------------------
__global__ void k_fill(float* __restrict__ p, float v, size_t n) {
    size_t i = (size_t)blockIdx.x * blockDim.x + threadIdx.x;
    if (i < n) p[i] = v;
}

// ---------------------------------------------------------------------------
// Weight convert + transpose: Wt[n][k] = (bf16) W[k][n] ; N fixed = 128
// ---------------------------------------------------------------------------
__global__ void k_w_to_bf16_t(const float* __restrict__ W, __bf16* __restrict__ Wt, int K) {
    int idx = blockIdx.x * blockDim.x + threadIdx.x;
    if (idx >= K * HN) return;
    int k = idx / HN, n = idx % HN;
    Wt[(size_t)n * K + k] = (__bf16)W[idx];
}

// ---------------------------------------------------------------------------
// Edge scatter-add (mean numerator + counts). One wave per edge.
// F is 128 or 256; lanes cover F/4 float4s.
// ---------------------------------------------------------------------------
__global__ void k_scatter_add(const float* __restrict__ xsrc, int F,
                              const int* __restrict__ src, const int* __restrict__ dst,
                              float* __restrict__ accum, float* __restrict__ cnt, int E) {
    int wavesPerBlk = blockDim.x >> 5;
    int e = blockIdx.x * wavesPerBlk + (threadIdx.x >> 5);
    int lane = threadIdx.x & 31;
    if (e >= E) return;
    int s = src[e];
    int d = dst[e];
    const float4* xs = (const float4*)(xsrc + (size_t)s * F);
    float* ap = accum + (size_t)d * F;
    int nvec = F >> 2;
    for (int j = lane; j < nvec; j += 32) {
        float4 v = xs[j];
        atomicAdd(ap + j * 4 + 0, v.x);
        atomicAdd(ap + j * 4 + 1, v.y);
        atomicAdd(ap + j * 4 + 2, v.z);
        atomicAdd(ap + j * 4 + 3, v.w);
    }
    if (lane == 0) atomicAdd(cnt + d, 1.0f);
}

__global__ void k_mean_div(float* __restrict__ accum, const float* __restrict__ cnt,
                           int M, int F) {
    size_t i = (size_t)blockIdx.x * blockDim.x + threadIdx.x;
    if (i >= (size_t)M * F) return;
    float c = cnt[i / F];
    accum[i] = accum[i] / fmaxf(c, 1.0f);
}

// ---------------------------------------------------------------------------
// WMMA GEMM: out[M x 128] = A[M x K] * W[K x 128] + bias
//   A   : f32, row-major, leading dim lda
//   Wt  : bf16, TRANSPOSED weights, Wt[n*K + k] = W[k][n]
//   out : f32, leading dim ldo (lets us write into concat halves)
// Block: 256 threads (8 waves) -> 128 rows x 128 cols. K staged via LDS, KC=64.
// Weight chunk staged by the Tensor Data Mover (one descriptor per chunk),
// A chunk staged by VALU (needs f32 -> bf16 convert).
// ---------------------------------------------------------------------------
__global__ void __launch_bounds__(256)
k_gemm_wmma(const float* __restrict__ A, int lda,
            const __bf16* __restrict__ Wt,
            const float* __restrict__ bias,
            float* __restrict__ out, int ldo,
            int M, int K) {
    __attribute__((aligned(16))) __shared__ __bf16 sA[128][KC];  // 16 KB
    __attribute__((aligned(16))) __shared__ __bf16 sW[128][KC];  // 16 KB

    const int tid  = threadIdx.x;
    const int wave = tid >> 5;
    const int lane = tid & 31;
    const int row0 = blockIdx.x * 128;

    const v8f vzero = {0.f, 0.f, 0.f, 0.f, 0.f, 0.f, 0.f, 0.f};
    v8f acc[8];
#pragma unroll
    for (int n = 0; n < 8; ++n) acc[n] = vzero;

    const int mrow = wave * 16 + (lane & 15);      // row within block tile
    const int ka   = (lane & 16) ? 8  : 0;         // A-frag K base (ISA 16-bit A layout)
    const int kb   = (lane & 16) ? 8 : 0;          // B-frag K base (ISA 16-bit B layout)
    const int ncl  = lane & 15;                    // column within 16-wide tile

    // LDS byte offset of sW: low 32 bits of the generic (shared-aperture) address
    const unsigned sw_lds = (unsigned)(unsigned long long)&sW[0][0];

    for (int k0 = 0; k0 < K; k0 += KC) {
#if HAVE_TDM
        // --- stage W chunk with the Tensor Data Mover ---------------------
        // D# (CDNA5 ISA): 2D tensor dim0=K, dim1=128 rows, stride0=K,
        // tile = KC x 128, data_size = 2 bytes, dest = sW (contiguous rows).
        if (tid == 0) {
            unsigned long long ga =
                (unsigned long long)(const void*)(Wt + (size_t)k0); // bytes
            u32x4 g0;
            g0.x = 1u;                                   // count=1, user desc
            g0.y = sw_lds;                               // lds_addr
            g0.z = (unsigned)(ga & 0xffffffffull);       // global_addr[31:0]
            g0.w = (unsigned)((ga >> 32) & 0x1ffffffull) // global_addr[56:32]
                 | 0x80000000u;                          // type=2 ("image")
            i32x8 g1;
            g1[0] = (1 << 16);                           // data_size=1 -> 2B
            g1[1] = (int)(((unsigned)K & 0xffffu) << 16);        // tensor_dim0 lo16
            g1[2] = (int)(((unsigned)K >> 16) | (128u << 16));   // dim0 hi | dim1 lo
            g1[3] = (int)(((unsigned)KC & 0xffffu) << 16);       // dim1 hi | tile_dim0
            g1[4] = 128;                                 // tile_dim1 (tile_dim2=0)
            g1[5] = K;                                   // tensor_dim0_stride lo32
            g1[6] = 0;                                   // stride0 hi | stride1 lo
            g1[7] = 0;
            i32x4 gz = {0, 0, 0, 0};
#if defined(__clang_major__) && __clang_major__ >= 23
            i32x8 gz8 = {0, 0, 0, 0, 0, 0, 0, 0};
            __builtin_amdgcn_tensor_load_to_lds(g0, g1, gz, gz, gz8, 0);
#else
            __builtin_amdgcn_tensor_load_to_lds(g0, g1, gz, gz, 0);
#endif
        }
#endif
        // --- stage A chunk (convert f32 -> bf16) --------------------------
        for (int f = tid; f < 128 * KC; f += 256) {
            int r = f >> 6, c = f & (KC - 1);
            int gr = row0 + r;
            float v = (gr < M) ? A[(size_t)gr * lda + k0 + c] : 0.0f;
            sA[r][c] = (__bf16)v;
        }
#if !HAVE_TDM
        for (int f = tid; f < 128 * KC; f += 256) {
            int n = f >> 6, c = f & (KC - 1);
            sW[n][c] = Wt[(size_t)n * K + k0 + c];
        }
#endif
        // prefetch next A chunk into cache hierarchy (global_prefetch_b8)
        if (k0 + KC < K) {
            int gr = row0 + (tid >> 1);
            if (gr < M)
                __builtin_prefetch(A + (size_t)gr * lda + k0 + KC + (tid & 1) * 32, 0, 1);
        }
#if HAVE_TDM
        if (tid == 0) {
#if __has_builtin(__builtin_amdgcn_s_wait_tensorcnt)
            __builtin_amdgcn_s_wait_tensorcnt(0);
#else
            asm volatile("s_wait_tensorcnt 0x0" ::: "memory");
#endif
        }
#endif
        __syncthreads();

        // --- compute: software-pipelined B-fragment loads -----------------
#pragma unroll
        for (int kk = 0; kk < KC; kk += 32) {
            V16U a;
            a.h[0] = *(const v8bf*)&sA[mrow][kk + ka];
            a.h[1] = *(const v8bf*)&sA[mrow][kk + 16 + ka];
            V16U bc, bn_;
            bc.h[0] = *(const v8bf*)&sW[ncl][kk + kb];
            bc.h[1] = *(const v8bf*)&sW[ncl][kk + kb + 16];
#pragma unroll
            for (int n = 0; n < 8; ++n) {
                if (n < 7) {
                    int ncol = (n + 1) * 16 + ncl;
                    bn_.h[0] = *(const v8bf*)&sW[ncol][kk + kb];
                    bn_.h[1] = *(const v8bf*)&sW[ncol][kk + kb + 16];
                }
                acc[n] = __builtin_amdgcn_wmma_f32_16x16x32_bf16(
                    /*neg_a=*/false, a.v, /*neg_b=*/false, bc.v,
                    /*c_mod=*/(short)0, acc[n],
                    /*reuse_a=*/false, /*reuse_b=*/false);
                bc = bn_;
            }
        }
        __syncthreads();
    }

    // --- epilogue: ISA 32-bit C/D layout: VGPR i -> row i + (lane>=16 ? 8:0) ---
    const int rbase = row0 + wave * 16 + ((lane & 16) ? 8 : 0);
#pragma unroll
    for (int n = 0; n < 8; ++n) {
        int col = n * 16 + ncl;
        float bv = bias[col];
#pragma unroll
        for (int i = 0; i < 8; ++i) {
            int gr = rbase + i;
            if (gr < M) out[(size_t)gr * ldo + col] = acc[n][i] + bv;
        }
    }
}

// ---------------------------------------------------------------------------
// BatchNorm stats: stat[0..127]=col sums, stat[128..255]=col sum of squares
// ---------------------------------------------------------------------------
__global__ void k_bn_stats(const float* __restrict__ x, int M, float* __restrict__ stat) {
    int c  = threadIdx.x;              // 128 threads = one per column
    int r0 = blockIdx.x * 64;
    int r1 = r0 + 64; if (r1 > M) r1 = M;
    float s = 0.f, q = 0.f;
    for (int r = r0; r < r1; ++r) {
        float v = x[(size_t)r * HN + c];
        s += v; q += v * v;
    }
    atomicAdd(&stat[c], s);
    atomicAdd(&stat[HN + c], q);
}

// BN(eps=1, batch stats, biased var) + LeakyReLU(0.01), in place
__global__ void k_bn_apply(float* __restrict__ x, int M, const float* __restrict__ stat,
                           const float* __restrict__ g, const float* __restrict__ be) {
    size_t i = (size_t)blockIdx.x * blockDim.x + threadIdx.x;
    if (i >= (size_t)M * HN) return;
    int c = (int)(i & (HN - 1));
    float invM = 1.0f / (float)M;
    float m = stat[c] * invM;
    float v = stat[HN + c] * invM - m * m;
    float y = g[c] * (x[i] - m) * rsqrtf(v + 1.0f) + be[c];
    x[i] = (y >= 0.f) ? y : 0.01f * y;
}

// ---------------------------------------------------------------------------
// Output head: out[i][l] = sum_k h[i][k] * w[k][l] + b[l], L <= 7
// ---------------------------------------------------------------------------
__global__ void k_head(const float* __restrict__ h, const float* __restrict__ w,
                       const float* __restrict__ b, float* __restrict__ out,
                       int M, int L) {
    int i = blockIdx.x * blockDim.x + threadIdx.x;
    if (i >= M) return;
    float acc[7];
    for (int l = 0; l < L; ++l) acc[l] = b[l];
    const float* hi = h + (size_t)i * HN;
    for (int k = 0; k < HN; ++k) {
        float x = hi[k];
        for (int l = 0; l < L; ++l) acc[l] += x * w[k * L + l];
    }
    for (int l = 0; l < L; ++l) out[(size_t)i * L + l] = acc[l];
}

// ---------------------------------------------------------------------------
// Host orchestration
// ---------------------------------------------------------------------------
extern "C" void kernel_launch(void* const* d_in, const int* in_sizes, int n_in,
                              void* d_out, int out_size, void* d_ws, size_t ws_size,
                              hipStream_t stream) {
    const int NA = 100000, NP = 100000;
    const int E  = in_sizes[2] / 2;

    // ---- inputs ----
    const float* x_author = (const float*)d_in[0];
    const float* x_paper  = (const float*)d_in[1];
    const int*   e_pa     = (const int*)d_in[2];   // [0:E)=paper src, [E:2E)=author dst
    const int*   e_ap     = (const int*)d_in[3];   // [0:E)=author src, [E:2E)=paper dst
    const int widx[12] = {4, 6, 8, 10, 12, 14, 16, 18, 20, 22, 24, 26};
    const int wK[12]   = {128, 256, 256, 256, 128, 256, 128, 128, 256, 128, 128, 256};
    const float* g1_a  = (const float*)d_in[28]; const float* be1_a = (const float*)d_in[29];
    const float* g1_p  = (const float*)d_in[30]; const float* be1_p = (const float*)d_in[31];
    const float* g2_a  = (const float*)d_in[32]; const float* be2_a = (const float*)d_in[33];
    const float* g2_p  = (const float*)d_in[34]; const float* be2_p = (const float*)d_in[35];
    const float* w_post_a = (const float*)d_in[36]; const float* b_post_a = (const float*)d_in[37];
    const float* w_post_p = (const float*)d_in[38]; const float* b_post_p = (const float*)d_in[39];
    float* out = (float*)d_out;

    // ---- workspace carve-up ----
    char* base = (char*)d_ws;
    size_t off = 0;
    auto alloc = [&](size_t bytes) -> char* {
        off = (off + 255) & ~(size_t)255;
        char* p = base + off;
        off += bytes;
        return p;
    };
    float* agg  = (float*)alloc((size_t)NP * 256 * 4);   // reused (F up to 256)
    float* cat  = (float*)alloc((size_t)NP * 256 * 4);   // [*, 2H] concat buffer (reused)
    float* h_a  = (float*)alloc((size_t)NA * HN * 4);
    float* h_p  = (float*)alloc((size_t)NP * HN * 4);
    float* h_a2 = (float*)alloc((size_t)NA * HN * 4);
    float* h_p2 = (float*)alloc((size_t)NP * HN * 4);
    float* cnt  = (float*)alloc((size_t)NA * 4);         // reused
    float* stat = (float*)alloc(256 * 4);                // reused per BN
    __bf16* wt[12];
    for (int i = 0; i < 12; ++i) wt[i] = (__bf16*)alloc((size_t)wK[i] * HN * 2);
    (void)ws_size; (void)n_in; (void)out_size;

    // ---- prep: weights -> bf16 transposed ----
    for (int i = 0; i < 12; ++i) {
        int tot = wK[i] * HN;
        k_w_to_bf16_t<<<(tot + 255) / 256, 256, 0, stream>>>(
            (const float*)d_in[widx[i]], wt[i], wK[i]);
    }

    auto fill = [&](float* p, size_t n) {
        k_fill<<<(unsigned)((n + 255) / 256), 256, 0, stream>>>(p, 0.f, n);
    };
    auto gemm = [&](const float* A, int lda, int wi, int bi, float* o, int ldo, int M, int K) {
        k_gemm_wmma<<<(M + 127) / 128, 256, 0, stream>>>(
            A, lda, wt[wi], (const float*)d_in[bi], o, ldo, M, K);
    };
    auto meanagg = [&](const float* xsrc, int F, const int* e, float* ac, int Mdst) {
        fill(cnt, (size_t)Mdst);
        fill(ac, (size_t)Mdst * F);
        k_scatter_add<<<(E + 7) / 8, 256, 0, stream>>>(xsrc, F, e, e + E, ac, cnt, E);
        k_mean_div<<<(unsigned)(((size_t)Mdst * F + 255) / 256), 256, 0, stream>>>(ac, cnt, Mdst, F);
    };
    auto bn = [&](float* x, int M, const float* g, const float* be) {
        fill(stat, 256);
        k_bn_stats<<<(M + 63) / 64, 128, 0, stream>>>(x, M, stat);
        k_bn_apply<<<(unsigned)(((size_t)M * HN + 255) / 256), 256, 0, stream>>>(x, M, stat, g, be);
    };

    // ================= layer 1 =================
    // author <- papers (p2a), src feats F=128
    meanagg(x_paper, 128, e_pa, agg, NA);
    gemm(x_author, 256, /*w1_p2a_dst*/1, 7,  cat,        256, NA, 256);
    gemm(agg,      128, /*w1_p2a_src*/0, 5,  cat + 128,  256, NA, 128);
    gemm(cat,      256, /*w1_p2a_upd*/2, 9,  h_a,        128, NA, 256);
    // paper <- authors (a2p), src feats F=256
    meanagg(x_author, 256, e_ap, agg, NP);
    gemm(x_paper, 128, /*w1_a2p_dst*/4, 13, cat,        256, NP, 128);
    gemm(agg,     256, /*w1_a2p_src*/3, 11, cat + 128,  256, NP, 256);
    gemm(cat,     256, /*w1_a2p_upd*/5, 15, h_p,        128, NP, 256);
    bn(h_a, NA, g1_a, be1_a);
    bn(h_p, NP, g1_p, be1_p);

    // ================= layer 2 =================
    // author <- papers, src feats = h_p (F=128)
    meanagg(h_p, 128, e_pa, agg, NA);
    gemm(h_a, 128, /*w2_p2a_dst*/7, 19, cat,       256, NA, 128);
    gemm(agg, 128, /*w2_p2a_src*/6, 17, cat + 128, 256, NA, 128);
    gemm(cat, 256, /*w2_p2a_upd*/8, 21, h_a2,      128, NA, 256);
    // paper <- authors, src feats = h_a (F=128)
    meanagg(h_a, 128, e_ap, agg, NP);
    gemm(h_p, 128, /*w2_a2p_dst*/10, 25, cat,       256, NP, 128);
    gemm(agg, 128, /*w2_a2p_src*/9,  23, cat + 128, 256, NP, 128);
    gemm(cat, 256, /*w2_a2p_upd*/11, 27, h_p2,      128, NP, 256);
    bn(h_a2, NA, g2_a, be2_a);
    bn(h_p2, NP, g2_p, be2_p);

    // ================= heads =================
    k_head<<<(NA + 255) / 256, 256, 0, stream>>>(h_a2, w_post_a, b_post_a, out, NA, 4);
    k_head<<<(NP + 255) / 256, 256, 0, stream>>>(h_p2, w_post_p, b_post_p, out + (size_t)NA * 4, NP, 7);
}
